// _LocalPassing_28896539968355
// MI455X (gfx1250) — matrix operand order repaired and checked by reading, not compiled, not run on hardware
//
#include <hip/hip_runtime.h>

typedef __attribute__((ext_vector_type(16))) _Float16 v16h;
typedef __attribute__((ext_vector_type(8)))  float    v8f;

#define CC  20      // classes
#define BB  2       // batch
#define HH  64
#define WW  2048
#define KN  24      // neighbors (5x5 minus center)
#define TS  16      // output tile (16x16)
#define TIN 20      // tile + 2-halo each side
#define DHS 32      // f16 data tile row stride (halves) -> 64B, 16B aligned, cols 20..31 zero (K pad)
#define NW  4       // waves per block
#define NT  128     // threads per block

union AFrag { v16h h; float4 f4[2]; };
union H8    { _Float16 h[8]; float4 f4; };

__global__ __launch_bounds__(NT, 2)   // >=2 waves/SIMD: cap VGPRs, kill s_set_vgpr_msb churn
void lp_kernel(const float* __restrict__ gData,
               const float* __restrict__ gMask,
               const float* __restrict__ gBil,
               const float* __restrict__ gAngW,
               const float* __restrict__ gBiW,
               float* __restrict__ gOutAng,
               float* __restrict__ gOutBi)
{
    __shared__ __align__(16) _Float16 sDataH[CC * TIN * DHS];  // 25600 B  f16 data tile (K-padded)
    __shared__ float    sXm  [CC * TIN * TIN];                 // 32000 B  f32 masked data tile
    __shared__ float    sMask[TIN * TIN];                      //  1600 B
    __shared__ float    sBil [KN * TS * TS];                   // 24576 B
    __shared__ _Float16 sTab [NW][2][64];                      //  1024 B  per-wave banded-weight LUT
    __shared__ __align__(16) _Float16 sT[NW][TS][32];          //  4096 B  T intermediate, col-major

    const int tid  = threadIdx.x;
    const int lane = tid & 31;
    const int wv   = tid >> 5;
    const int m16  = lane & 15;
    const int hi   = lane >> 4;

    const int h0 = blockIdx.y * TS;
    const int w0 = blockIdx.x * TS;
    const int bb = blockIdx.z;

    // ---- stage mask tile (with halo, zero-padded) ----
    for (int t = tid; t < TIN * TIN; t += NT) {
        int r = t / TIN, cl = t % TIN;
        int gh = h0 - 2 + r, gw = w0 - 2 + cl;
        float m = 0.f;
        if (gh >= 0 && gh < HH && gw >= 0 && gw < WW)
            m = gMask[(size_t)(bb * HH + gh) * WW + gw];
        sMask[t] = m;
    }
    __syncthreads();

    // ---- stage data tile: f16 (K-padded to 32 cols) + f32 xm = data*mask ----
    for (int t = tid; t < CC * TIN * DHS; t += NT) {
        int c   = t / (TIN * DHS);
        int rem = t % (TIN * DHS);
        int r = rem / DHS, cl = rem % DHS;
        float dv = 0.f;
        if (cl < TIN) {
            int gh = h0 - 2 + r, gw = w0 - 2 + cl;
            if (gh >= 0 && gh < HH && gw >= 0 && gw < WW)
                dv = gData[((size_t)(bb * CC + c) * HH + gh) * WW + gw];
            sXm[(c * TIN + r) * TIN + cl] = dv * sMask[r * TIN + cl];
        }
        sDataH[t] = (_Float16)dv;
    }
    // ---- stage bilateral tile (interior only, reused by all 20 channels) ----
    for (int t = tid; t < KN * TS * TS; t += NT) {
        int j = t / (TS * TS), p = t % (TS * TS);
        int r = p / TS, cl = p % TS;
        sBil[t] = gBil[((size_t)(bb * KN + j) * HH + (h0 + r)) * WW + (w0 + cl)];
    }
    __syncthreads();

    // ---- per-wave channel loop: wave wv owns channels wv, wv+4, ... ----
    for (int cv = wv; cv < CC; cv += NW) {
        // c is wave-uniform by construction: pin it to an SGPR so all the gather
        // index math (divisions by 20, neighbor offsets) lowers to SALU, once per channel.
        const int c = __builtin_amdgcn_readfirstlane(cv);

        // Precompute the 24 scalar xm offsets for the torch-view gather:
        // f = c*24+j -> neighbor f/20 (center removed), channel f%20.
        int offXm[KN];
        #pragma unroll
        for (int j = 0; j < KN; ++j) {
            int f   = c * KN + j;
            int nb  = f / 20;
            int ch  = f - nb * 20;
            int idx = nb + (nb >= 12 ? 1 : 0);
            int di  = idx / 5 - 2, dj = idx % 5 - 2;
            offXm[j] = (ch * TIN + 2 + di) * TIN + (2 + dj);
        }

        // Separable factors from kernel row i=2: k[2][j] = g[2]*g[j] = g[j]; g[2]=1.
        // LUT tab[idx] = (idx-15 in [0,4]) ? g[idx-15] : 0, so band value = tab[k - n + 15].
        for (int i = lane; i < 64; i += 32) {
            int t5 = i - 15;
            float ga = 0.f, gb = 0.f;
            if (t5 >= 0 && t5 <= 4) {
                ga = (t5 == 2) ? 1.0f : gAngW[c * 25 + 10 + t5];
                gb = (t5 == 2) ? 1.0f : gBiW [c * 25 + 10 + t5];
            }
            sTab[wv][0][i] = (_Float16)ga;
            sTab[wv][1][i] = (_Float16)gb;
        }
        // in-wave LDS store->load is in-order (DScnt); wave-private buffers, no barrier needed

        // A fragments (X tile rows), shared by both convs.
        // f16 A 16x32 layout: lanes0-15 row=m16 K{0..7,16..23}; lanes16-31 row=m16 K{8..15,24..31}
        AFrag a0, a1;
        {
            const _Float16* row = &sDataH[(c * TIN + m16) * DHS];
            a0.f4[0] = *(const float4*)&row[hi * 8];
            a0.f4[1] = *(const float4*)&row[16 + hi * 8];
            a1.f4[0] = make_float4(0.f, 0.f, 0.f, 0.f);
            a1.f4[1] = make_float4(0.f, 0.f, 0.f, 0.f);
            if (m16 < 4) {  // T1 tile covers X rows 16..19; rows >=4 stay zero
                const _Float16* row2 = &sDataH[(c * TIN + 16 + m16) * DHS];
                a1.f4[0] = *(const float4*)&row2[hi * 8];
                a1.f4[1] = *(const float4*)&row2[16 + hi * 8];
            }
        }
        // f16-rounded center values (exactly cancel the g[2]^2=1 center tap of the separable conv)
        float ctr[8];
        #pragma unroll
        for (int v = 0; v < 8; ++v)
            ctr[v] = (float)sDataH[(c * TIN + (v + 8 * hi) + 2) * DHS + (m16 + 2)];

        v8f yAng = {}, yBi = {};
        #pragma unroll
        for (int pass = 0; pass < 2; ++pass) {
            const _Float16* tab = sTab[wv][pass];

            // B = Wh (32x16 band): lane holds col n=m16, K{hi*8+0..7, 16+hi*8+0..7}
            v16h bh;
            {
                int base = hi * 8 - m16 + 15;
                #pragma unroll
                for (int e = 0; e < 16; ++e) bh[e] = tab[base + e + ((e < 8) ? 0 : 8)];
            }
            // pass 1: T = X * Wh   (two 16x16 D tiles: rows 0..15 and 16..19)
            v8f t0 = {}, t1 = {};
            t0 = __builtin_amdgcn_wmma_f32_16x16x32_f16(false, a0.h, false, bh, (short)0, t0, false, false);
            t1 = __builtin_amdgcn_wmma_f32_16x16x32_f16(false, a1.h, false, bh, (short)0, t1, false, false);

            // store T to LDS column-major sT[col][rowK]; each lane's 8 rows are contiguous
            // halves -> pack (v_cvt_pk_f16_f32) and emit one ds_store_b128 per tile.
            H8 p0, p1;
            #pragma unroll
            for (int v = 0; v < 8; ++v) {
                p0.h[v] = (_Float16)t0[v];   // T row 8*hi + v
                p1.h[v] = (_Float16)t1[v];   // T row 16 + 8*hi + v (rows >=20 are true zeros)
            }
            *(float4*)&sT[wv][m16][8 * hi]      = p0.f4;
            *(float4*)&sT[wv][m16][16 + 8 * hi] = p1.f4;

            // pass 2: Y = Wv * T.  A = Wv band (lane row m=m16), B = T columns from sT
            v16h av;
            #pragma unroll
            for (int e = 0; e < 16; ++e) {
                int k = (e < 8) ? (hi * 8 + e) : (16 + hi * 8 + (e - 8));
                av[e] = tab[k - m16 + 15];
            }
            AFrag bt;
            bt.f4[0] = *(const float4*)&sT[wv][m16][hi * 8];
            bt.f4[1] = *(const float4*)&sT[wv][m16][16 + hi * 8];

            v8f y = {};
            y = __builtin_amdgcn_wmma_f32_16x16x32_f16(false, av, false, bt.h, (short)0, y, false, false);
            if (pass == 0) yAng = y; else yBi = y;
        }

        // ---- outputs: ang, and gather * mask * bi_ang ----
        #pragma unroll
        for (int v = 0; v < 8; ++v) {
            int row = v + 8 * hi, col = m16;
            int vbase = row * TIN + col;            // per-lane xm base (element units)
            float ang = yAng[v] - ctr[v];
            float biA = yBi[v]  - ctr[v];

            float s = 0.f;
            #pragma unroll
            for (int j = 0; j < KN; ++j) {
                s += sBil[(j * TS + row) * TS + col] * sXm[offXm[j] + vbase];
            }
            float mk = sMask[(row + 2) * TIN + (col + 2)];
            size_t o = ((size_t)(bb * CC + c) * HH + (h0 + row)) * WW + (w0 + col);
            gOutAng[o] = ang;
            gOutBi[o]  = s * mk * biA;
        }
    }
}

extern "C" void kernel_launch(void* const* d_in, const int* in_sizes, int n_in,
                              void* d_out, int out_size, void* d_ws, size_t ws_size,
                              hipStream_t stream) {
    const float* data = (const float*)d_in[0];
    const float* mask = (const float*)d_in[1];
    const float* bil  = (const float*)d_in[2];
    const float* aw   = (const float*)d_in[3];
    const float* bw   = (const float*)d_in[4];
    float* outAng = (float*)d_out;
    float* outBi  = outAng + (size_t)BB * CC * HH * WW;
    dim3 grid(WW / TS, HH / TS, BB);
    lp_kernel<<<grid, dim3(NT), 0, stream>>>(data, mask, bil, aw, bw, outAng, outBi);
}
